// Transformer_43636867727564
// MI455X (gfx1250) — matrix operand order repaired and checked by reading, not compiled, not run on hardware
//
#include <hip/hip_runtime.h>
#include <math.h>

#define B_  4
#define S_  160
#define D_  512
#define H_  8
#define L_  2
#define FF_ 2048
#define WD_ 64
#define M_  (B_*S_)   // 640 rows
#define LDP 72        // padded LDS row stride (bf16 elems), 144B = 9*16B

typedef __attribute__((ext_vector_type(16))) __bf16 v16bf;
typedef __attribute__((ext_vector_type(8)))  float  v8f;
typedef __attribute__((vector_size(16)))     int    v4i_t;

#if defined(__HIP_DEVICE_COMPILE__) && __has_builtin(__builtin_amdgcn_global_load_async_to_lds_b128)
#define HAVE_ASYNC 1
#else
#define HAVE_ASYNC 0
#endif

__device__ __forceinline__ __bf16 to_bf16(float f) {
    unsigned int u = __float_as_uint(f);
    unsigned int r = (u + 0x7FFFu + ((u >> 16) & 1u)) >> 16;
    unsigned short hs = (unsigned short)r;
    return __builtin_bit_cast(__bf16, hs);
}

// copy 16 bytes global -> LDS (async DMA if available, else regular ld/st)
__device__ __forceinline__ void cp16_g2l(const __bf16* g, __bf16* l) {
#if HAVE_ASYNC
    __builtin_amdgcn_global_load_async_to_lds_b128((v4i_t*)g, (v4i_t*)l, 0, 0);
#else
    *(uint4*)l = *(const uint4*)g;
#endif
}

__device__ __forceinline__ void wait_async_lds() {
#if HAVE_ASYNC
#if __has_builtin(__builtin_amdgcn_s_wait_asynccnt)
    __builtin_amdgcn_s_wait_asynccnt(0);
#else
    asm volatile("s_wait_asynccnt 0" ::: "memory");
#endif
#endif
}

// ---------------- fp32 -> bf16 conversion ----------------
__global__ void k_cvt_bf16(const float* __restrict__ src, __bf16* __restrict__ dst, int n) {
    int i = blockIdx.x * blockDim.x + threadIdx.x;
    int stride = gridDim.x * blockDim.x;
    for (; i < n; i += stride) dst[i] = to_bf16(src[i]);
}

// ---------------- fp32 [K,N] -> bf16 transposed [N,K] ----------------
__global__ void k_cvt_bf16_T(const float* __restrict__ src, __bf16* __restrict__ dst,
                             int K, int N) {
    size_t total = (size_t)K * N;
    size_t stride = (size_t)gridDim.x * blockDim.x;
    for (size_t i = (size_t)blockIdx.x * blockDim.x + threadIdx.x; i < total; i += stride) {
        int n = (int)(i / K);
        int k = (int)(i % K);
        dst[i] = to_bf16(src[(size_t)k * N + n]);   // dst[n*K + k]
    }
}

// ---------------- WMMA GEMM: C[M,N] = A[M,K](bf16) @ Bt[N,K](bf16, pre-transposed) + bias --
// 256 threads = 8 waves (2x4 wave grid); block tile 32x64; K staged 64/stage, double-buffered.
// grid = (N/64, M/32). M,N,K divide evenly for this problem.
__global__ __launch_bounds__(256) void k_gemm_bf16(
    const __bf16* __restrict__ A, const __bf16* __restrict__ Bt,
    const float* __restrict__ bias, float* __restrict__ C,
    int Mdim, int Ndim, int Kdim) {
    __shared__ __bf16 sh_a[2][32 * LDP];   // [row][k]
    __shared__ __bf16 sh_b[2][64 * LDP];   // [col][k]

    const int tid  = threadIdx.x;
    const int wave = tid >> 5;
    const int lane = tid & 31;
    const int g    = lane >> 4;
    const int r    = lane & 15;
    const int wm   = wave >> 2;   // 0..1 -> 16-row slab
    const int wn   = wave & 3;    // 0..3 -> 16-col slab
    const int rowBase = blockIdx.y * 32;
    const int colBase = blockIdx.x * 64;

    // per-thread copy coordinates (16-byte chunks along K)
    const int arow = tid >> 3;                          // A: 32 rows x 8 chunks
    const int bn0  = tid >> 3, bn1 = (tid + 256) >> 3;  // B: 64 cols x 8 chunks (2/thread)
    const int kc   = (tid & 7) * 8;

    const __bf16* aptr  = A  + (size_t)(rowBase + arow) * Kdim + kc;
    const __bf16* bptr0 = Bt + (size_t)(colBase + bn0) * Kdim + kc;
    const __bf16* bptr1 = Bt + (size_t)(colBase + bn1) * Kdim + kc;

    v8f acc = {};
    const int nstage = Kdim >> 6;

    // prologue: stage 0
    cp16_g2l(aptr,  &sh_a[0][arow * LDP + kc]);
    cp16_g2l(bptr0, &sh_b[0][bn0  * LDP + kc]);
    cp16_g2l(bptr1, &sh_b[0][bn1  * LDP + kc]);
    wait_async_lds();
    __syncthreads();

    for (int st = 0; st < nstage; ++st) {
        const int cur = st & 1, nxt = cur ^ 1;
        // issue next stage DMA while computing current stage; buffer `nxt` was
        // released by the barrier at the end of stage st-1.
        if (st + 1 < nstage) {
            const int koff = (st + 1) << 6;
            cp16_g2l(aptr  + koff, &sh_a[nxt][arow * LDP + kc]);
            cp16_g2l(bptr0 + koff, &sh_b[nxt][bn0  * LDP + kc]);
            cp16_g2l(bptr1 + koff, &sh_b[nxt][bn1  * LDP + kc]);
        }
        if (st + 2 < nstage) {  // prefetch stage+2 into caches
            const int koff = (st + 2) << 6;
            __builtin_prefetch(aptr + koff, 0, 1);
            __builtin_prefetch(bptr0 + koff, 0, 1);
        }

        // ---- two 16x16x32 WMMAs from LDS ----
#pragma unroll
        for (int kk = 0; kk < 2; ++kk) {
            const int kof = kk * 32;
            v16bf af, bfv;
            // A 16x32 frag: lane holds row (lane&15); K split by lane group
#pragma unroll
            for (int vv = 0; vv < 8; ++vv) {
                int kb = kof + 8 * g + ((vv < 4) ? (2 * vv) : (16 + 2 * (vv - 4)));
                af[2 * vv]     = sh_a[cur][(wm * 16 + r) * LDP + kb];
                af[2 * vv + 1] = sh_a[cur][(wm * 16 + r) * LDP + kb + 1];
            }
            // B 32x16 frag from [col][k] tile: lane holds col; K split by lane group (same as A)
#pragma unroll
            for (int vv = 0; vv < 8; ++vv) {
                int kkk = kof + 8 * g + ((vv < 4) ? (2 * vv) : (16 + 2 * (vv - 4)));
                bfv[2 * vv]     = sh_b[cur][(wn * 16 + r) * LDP + kkk];
                bfv[2 * vv + 1] = sh_b[cur][(wn * 16 + r) * LDP + kkk + 1];
            }
            acc = __builtin_amdgcn_wmma_f32_16x16x32_bf16(
                false, af, false, bfv, (short)0, acc, false, false);
        }
        wait_async_lds();
        __syncthreads();
    }

    const float bb = bias ? bias[colBase + wn * 16 + r] : 0.0f;
#pragma unroll
    for (int vv = 0; vv < 8; ++vv) {
        int m = rowBase + wm * 16 + 8 * g + vv;
        C[(size_t)m * Ndim + (colBase + wn * 16 + r)] = acc[vv] + bb;
    }
}

// ---------------- embedding + LayerNorm ----------------
__global__ __launch_bounds__(256) void k_embed_ln(
    const int* __restrict__ x, const int* __restrict__ stamp,
    const float* __restrict__ tok_emb, const float* __restrict__ month_emb,
    const float* __restrict__ day_emb, const float* __restrict__ g0,
    const float* __restrict__ b0, float* __restrict__ h, __bf16* __restrict__ hb) {
    const int row = blockIdx.x;            // b*S + s
    const int b = row / S_, s = row % S_;
    const int tid = threadIdx.x;
    __shared__ float red[256];

    const float* tk = tok_emb + (size_t)x[row] * D_;
    float v0, v1;
    if (s > 0) {
        const int* st = stamp + (size_t)(b * (S_ - 1) + (s - 1)) * 3;
        const float* tm = month_emb + (size_t)st[0] * D_;
        const float* td = day_emb   + (size_t)st[1] * D_;
        v0 = tk[tid]       + tm[tid]       + td[tid];
        v1 = tk[tid + 256] + tm[tid + 256] + td[tid + 256];
    } else {
        v0 = tk[tid];
        v1 = tk[tid + 256];
    }
    red[tid] = v0 + v1; __syncthreads();
    for (int o = 128; o > 0; o >>= 1) { if (tid < o) red[tid] += red[tid + o]; __syncthreads(); }
    float mean = red[0] / (float)D_; __syncthreads();
    float d0 = v0 - mean, d1 = v1 - mean;
    red[tid] = d0 * d0 + d1 * d1; __syncthreads();
    for (int o = 128; o > 0; o >>= 1) { if (tid < o) red[tid] += red[tid + o]; __syncthreads(); }
    float inv = rsqrtf(red[0] / (float)D_ + 1e-12f);
    float o0 = g0[tid] * d0 * inv + b0[tid];
    float o1 = g0[tid + 256] * d1 * inv + b0[tid + 256];
    size_t base = (size_t)row * D_;
    h[base + tid] = o0;            h[base + tid + 256] = o1;
    hb[base + tid] = to_bf16(o0);  hb[base + tid + 256] = to_bf16(o1);
}

// ---------------- residual add + LayerNorm ----------------
__global__ __launch_bounds__(256) void k_add_ln(
    const float* __restrict__ xin, const float* __restrict__ res,
    const float* __restrict__ gamma, const float* __restrict__ beta,
    float* __restrict__ outf, __bf16* __restrict__ outb) {
    const int row = blockIdx.x;
    const int tid = threadIdx.x;
    __shared__ float red[256];
    size_t base = (size_t)row * D_;
    float v0 = xin[base + tid]       + res[base + tid];
    float v1 = xin[base + tid + 256] + res[base + tid + 256];
    red[tid] = v0 + v1; __syncthreads();
    for (int o = 128; o > 0; o >>= 1) { if (tid < o) red[tid] += red[tid + o]; __syncthreads(); }
    float mean = red[0] / (float)D_; __syncthreads();
    float d0 = v0 - mean, d1 = v1 - mean;
    red[tid] = d0 * d0 + d1 * d1; __syncthreads();
    for (int o = 128; o > 0; o >>= 1) { if (tid < o) red[tid] += red[tid + o]; __syncthreads(); }
    float inv = rsqrtf(red[0] / (float)D_ + 1e-12f);
    float o0 = gamma[tid] * d0 * inv + beta[tid];
    float o1 = gamma[tid + 256] * d1 * inv + beta[tid + 256];
    outf[base + tid] = o0;            outf[base + tid + 256] = o1;
    outb[base + tid] = to_bf16(o0);   outb[base + tid + 256] = to_bf16(o1);
}

// ---------------- attention with temporal K/V gathers ----------------
__global__ __launch_bounds__(256) void k_attn(
    const float* __restrict__ q, const float* __restrict__ k,
    const float* __restrict__ v, const int* __restrict__ time_matrix,
    const int* __restrict__ mask, const float* __restrict__ tiK,
    const float* __restrict__ tiV, float* __restrict__ hh) {
    const int blk = blockIdx.x;
    const int b  = blk / (S_ * H_);
    const int t  = blk % (S_ * H_);
    const int i  = t / H_;
    const int hd = t % H_;
    const int tid = threadIdx.x;
    __shared__ float q_sh[WD_];
    __shared__ float sc[S_];
    __shared__ int   tm_sh[S_];
    __shared__ float red[256];

    const size_t qoff = (size_t)(b * S_ + i) * D_ + hd * WD_;
    if (tid < WD_) q_sh[tid] = q[qoff + tid];
    __syncthreads();

    if (tid < S_) {
        int tm = time_matrix[(size_t)(b * S_ + i) * S_ + tid];
        tm_sh[tid] = tm;
        const float4* kr4  = (const float4*)(k   + (size_t)(b * S_ + tid) * D_ + hd * WD_);
        const float4* tk4  = (const float4*)(tiK + (size_t)tm * D_ + hd * WD_);
        float s = 0.f;
#pragma unroll
        for (int w = 0; w < WD_ / 4; ++w) {
            float4 a = kr4[w], c = tk4[w];
            s += q_sh[4*w]   * (a.x + c.x);
            s += q_sh[4*w+1] * (a.y + c.y);
            s += q_sh[4*w+2] * (a.z + c.z);
            s += q_sh[4*w+3] * (a.w + c.w);
        }
        float mf = (float)mask[b * S_ + tid];
        sc[tid] = s * 0.125f + 10000.0f * (1.0f - mf);
    }
    __syncthreads();
    red[tid] = (tid < S_) ? sc[tid] : -3.4e38f;
    __syncthreads();
    for (int o = 128; o > 0; o >>= 1) { if (tid < o) red[tid] = fmaxf(red[tid], red[tid + o]); __syncthreads(); }
    float mx = red[0]; __syncthreads();
    float e = 0.f;
    if (tid < S_) e = expf(sc[tid] - mx);
    red[tid] = e; __syncthreads();
    for (int o = 128; o > 0; o >>= 1) { if (tid < o) red[tid] += red[tid + o]; __syncthreads(); }
    float inv = 1.0f / red[0]; __syncthreads();
    if (tid < S_) sc[tid] = e * inv;
    __syncthreads();
    if (tid < WD_) {
        float acc = 0.f;
        for (int j = 0; j < S_; ++j) {
            float p = sc[j];
            acc += p * (v[(size_t)(b * S_ + j) * D_ + hd * WD_ + tid]
                      + tiV[(size_t)tm_sh[j] * D_ + hd * WD_ + tid]);
        }
        hh[(size_t)(b * S_ + i) * D_ + hd * WD_ + tid] = acc;
    }
}

// ---------------- exact GELU + bf16 cast ----------------
__global__ void k_gelu_bf16(const float* __restrict__ xin, __bf16* __restrict__ outb, int n) {
    int i = blockIdx.x * blockDim.x + threadIdx.x;
    int stride = gridDim.x * blockDim.x;
    for (; i < n; i += stride) {
        float x = xin[i];
        float gl = 0.5f * x * (1.0f + erff(x * 0.70710678118654752f));
        outb[i] = to_bf16(gl);
    }
}

extern "C" void kernel_launch(void* const* d_in, const int* in_sizes, int n_in,
                              void* d_out, int out_size, void* d_ws, size_t ws_size,
                              hipStream_t stream) {
    const int*   x       = (const int*)d_in[0];
    const int*   stamp   = (const int*)d_in[1];
    const int*   mask    = (const int*)d_in[2];
    const int*   tmat    = (const int*)d_in[3];
    const float* tok_emb = (const float*)d_in[4];
    const float* mon_emb = (const float*)d_in[5];
    const float* day_emb = (const float*)d_in[6];
    const float* tiK     = (const float*)d_in[7];
    const float* tiV     = (const float*)d_in[8];
    const float* g0 = (const float*)d_in[9];
    const float* b0 = (const float*)d_in[10];
    const float* Wq = (const float*)d_in[11]; const float* bq = (const float*)d_in[12];
    const float* Wk = (const float*)d_in[13]; const float* bk = (const float*)d_in[14];
    const float* Wv = (const float*)d_in[15]; const float* bv = (const float*)d_in[16];
    const float* Wo = (const float*)d_in[17]; const float* bo = (const float*)d_in[18];
    const float* g1 = (const float*)d_in[19]; const float* b1 = (const float*)d_in[20];
    const float* W1 = (const float*)d_in[21]; const float* c1 = (const float*)d_in[22];
    const float* W2 = (const float*)d_in[23]; const float* c2 = (const float*)d_in[24];
    const float* g2 = (const float*)d_in[25]; const float* b2 = (const float*)d_in[26];

    char* p = (char*)d_ws;
    auto alloc = [&](size_t bytes) -> void* {
        void* r = (void*)p;
        p += (bytes + 255) & ~(size_t)255;
        return r;
    };
    const size_t WDD = (size_t)L_ * D_ * D_;
    const size_t WDF = (size_t)L_ * D_ * FF_;
    __bf16* wq_bf = (__bf16*)alloc(WDD * 2);   // transposed [N][K] per layer
    __bf16* wk_bf = (__bf16*)alloc(WDD * 2);
    __bf16* wv_bf = (__bf16*)alloc(WDD * 2);
    __bf16* wo_bf = (__bf16*)alloc(WDD * 2);
    __bf16* w1_bf = (__bf16*)alloc(WDF * 2);   // [FF][D] per layer
    __bf16* w2_bf = (__bf16*)alloc(WDF * 2);   // [D][FF] per layer
    float*  h    = (float*)alloc((size_t)M_ * D_ * 4);
    __bf16* hb   = (__bf16*)alloc((size_t)M_ * D_ * 2);
    float*  qb   = (float*)alloc((size_t)M_ * D_ * 4);
    float*  kb   = (float*)alloc((size_t)M_ * D_ * 4);
    float*  vb   = (float*)alloc((size_t)M_ * D_ * 4);
    float*  hhf  = (float*)alloc((size_t)M_ * D_ * 4);
    __bf16* hhb  = (__bf16*)alloc((size_t)M_ * D_ * 2);
    float*  ao   = (float*)alloc((size_t)M_ * D_ * 4);
    float*  ff1  = (float*)alloc((size_t)M_ * FF_ * 4);
    __bf16* ff1b = (__bf16*)alloc((size_t)M_ * FF_ * 2);
    float*  ff2  = (float*)alloc((size_t)M_ * D_ * 4);
    (void)ws_size; (void)n_in; (void)in_sizes; (void)out_size;

    auto cvtT = [&](const float* src, __bf16* dst, int K, int N) {
        size_t n = (size_t)K * N;
        int blocks = (int)((n + 255) / 256);
        if (blocks > 2048) blocks = 2048;
        k_cvt_bf16_T<<<blocks, 256, 0, stream>>>(src, dst, K, N);
    };
    auto cvt = [&](const float* src, __bf16* dst, size_t n) {
        int blocks = (int)((n + 255) / 256);
        if (blocks > 2048) blocks = 2048;
        k_cvt_bf16<<<blocks, 256, 0, stream>>>(src, dst, (int)n);
    };
    for (int l = 0; l < L_; ++l) {
        const size_t odd = (size_t)l * D_ * D_;
        const size_t odf = (size_t)l * D_ * FF_;
        cvtT(Wq + odd, wq_bf + odd, D_, D_);
        cvtT(Wk + odd, wk_bf + odd, D_, D_);
        cvtT(Wv + odd, wv_bf + odd, D_, D_);
        cvtT(Wo + odd, wo_bf + odd, D_, D_);
        cvtT(W1 + odf, w1_bf + odf, D_, FF_);
        cvtT(W2 + odf, w2_bf + odf, FF_, D_);
    }

    k_embed_ln<<<M_, 256, 0, stream>>>(x, stamp, tok_emb, mon_emb, day_emb, g0, b0, h, hb);

    dim3 gD(D_ / 64, M_ / 32);    // (8,20)  for N=D
    dim3 gF(FF_ / 64, M_ / 32);   // (32,20) for N=FF

    for (int l = 0; l < L_; ++l) {
        const size_t odd = (size_t)l * D_ * D_;
        const size_t odf = (size_t)l * D_ * FF_;
        k_gemm_bf16<<<gD, 256, 0, stream>>>(hb, wq_bf + odd, bq + (size_t)l * D_, qb, M_, D_, D_);
        k_gemm_bf16<<<gD, 256, 0, stream>>>(hb, wk_bf + odd, bk + (size_t)l * D_, kb, M_, D_, D_);
        k_gemm_bf16<<<gD, 256, 0, stream>>>(hb, wv_bf + odd, bv + (size_t)l * D_, vb, M_, D_, D_);
        k_attn<<<B_ * S_ * H_, 256, 0, stream>>>(qb, kb, vb, tmat, mask, tiK, tiV, hhf);
        cvt(hhf, hhb, (size_t)M_ * D_);
        k_gemm_bf16<<<gD, 256, 0, stream>>>(hhb, wo_bf + odd, bo + (size_t)l * D_, ao, M_, D_, D_);
        k_add_ln<<<M_, 256, 0, stream>>>(h, ao, g1 + (size_t)l * D_, b1 + (size_t)l * D_, h, hb);
        k_gemm_bf16<<<gF, 256, 0, stream>>>(hb, w1_bf + odf, c1 + (size_t)l * FF_, ff1, M_, FF_, D_);
        {
            int n = M_ * FF_;
            int blocks = (n + 255) / 256; if (blocks > 2048) blocks = 2048;
            k_gelu_bf16<<<blocks, 256, 0, stream>>>(ff1, ff1b, n);
        }
        k_gemm_bf16<<<gD, 256, 0, stream>>>(ff1b, w2_bf + odf, c2 + (size_t)l * D_, ff2, M_, D_, FF_);
        float* outf = (l == L_ - 1) ? (float*)d_out : h;
        k_add_ln<<<M_, 256, 0, stream>>>(h, ff2, g2 + (size_t)l * D_, b2 + (size_t)l * D_, outf, hb);
    }
}
